// Multimodal_SelfAttention_21071109554307
// MI455X (gfx1250) — compile-verified, matrix operand order checked by reading, not run on hardware
//
#include <hip/hip_runtime.h>
#include <cstdint>
#include <cstddef>

#define BB  256
#define SS  200
#define HH  768
#define NHH 12
#define HDD 64
#define MM  (BB*SS)   // 51200 rows

typedef __attribute__((ext_vector_type(8)))  float  v8f;
typedef __attribute__((ext_vector_type(16))) __bf16 v16bf;
typedef __attribute__((ext_vector_type(4)))  unsigned int u32x4;
typedef __attribute__((ext_vector_type(8)))  int i32x8;
typedef __attribute__((ext_vector_type(4)))  int i32x4;

#if __has_builtin(__builtin_amdgcn_tensor_load_to_lds)
#define HAVE_TDM 1
#else
#define HAVE_TDM 0
#endif

// ---------------- helpers ----------------

__device__ __forceinline__ __bf16 f2bf(float f) {
  union { float f; uint32_t u; } x; x.f = f;
  uint32_t r = x.u + 0x7FFFu + ((x.u >> 16) & 1u);  // round-to-nearest-even
  union { unsigned short s; __bf16 b; } y; y.s = (unsigned short)(r >> 16);
  return y.b;
}

// 16-bit A/B fragment for V_WMMA_F32_16X16X32_BF16 from a row-major bf16 LDS
// tile. Lane L holds row L%16; K-groups {(L/16)*8..+7} and {16+(L/16)*8..+7}.
__device__ __forceinline__ v16bf ldfrag(const __bf16* base, int stride, int row,
                                        int kb, int hi) {
  union { uint4 u[2]; v16bf v; } f;
  const __bf16* p = base + (size_t)row * stride + kb + hi * 8;
  f.u[0] = *(const uint4*)p;
  f.u[1] = *(const uint4*)(p + 16);
  return f.v;
}

// Same fragment but sourced from an f32 LDS tile (probabilities), converting.
__device__ __forceinline__ v16bf ldfrag_f32(const float* base, int stride,
                                            int row, int kb, int hi) {
  union { __bf16 e[16]; v16bf v; } f;
  const float* p = base + (size_t)row * stride + kb + hi * 8;
#pragma unroll
  for (int i = 0; i < 8; i++) f.e[i] = f2bf(p[i]);
#pragma unroll
  for (int i = 0; i < 8; i++) f.e[8 + i] = f2bf(p[16 + i]);
  return f.v;
}

// gfx1250 async global->LDS 16B copy (tracked by ASYNCcnt)
__device__ __forceinline__ void async_ld16(uint32_t ldsOff, const void* gptr) {
  asm volatile("global_load_async_to_lds_b128 %0, %1, off"
               :: "v"(ldsOff), "v"(gptr) : "memory");
}
__device__ __forceinline__ void wait_async(int pending) {
  if (pending == 0)      asm volatile("s_wait_asynccnt 0x0" ::: "memory");
  else if (pending == 2) asm volatile("s_wait_asynccnt 0x2" ::: "memory");
  else                   asm volatile("s_wait_asynccnt 0x3" ::: "memory");
}

// TDM: load a 64-row x 32-col bf16 tile (row stride 768 elems) into LDS.
// D# layout per CDNA5 ISA ch.8 (group0: count/lds/global/type, group1: dims).
__device__ __forceinline__ void tdm_load_tile(uint32_t ldsOff,
                                              const __bf16* gptr) {
#if HAVE_TDM
  uint64_t ga  = (uint64_t)(uintptr_t)gptr;
  uint32_t l   = __builtin_amdgcn_readfirstlane(ldsOff);
  uint32_t glo = __builtin_amdgcn_readfirstlane((uint32_t)ga);
  uint32_t ghi = __builtin_amdgcn_readfirstlane((uint32_t)(ga >> 32));
  u32x4 g0 = { 1u,                                   // count=1, user desc
               l,                                    // lds_addr
               glo,                                  // global_addr[31:0]
               (ghi & 0x1FFFFFFu) | (2u << 30) };    // ga[56:32] | type=2
  i32x8 g1 = { (int)(1u << 16),                      // data_size = 2 bytes
               (int)((768u & 0xFFFFu) << 16),        // tensor_dim0 lo16
               (int)((768u >> 16) | ((768u & 0xFFFFu) << 16)), // d0 hi/d1 lo
               (int)((768u >> 16) | (32u << 16)),    // d1 hi | tile_dim0=32
               (int)64,                              // tile_dim1=64, dim2=0
               (int)768,                             // dim0_stride lo32
               0, 0 };                               // stride hi / d1 stride
  i32x4 z4 = { 0, 0, 0, 0 };
  i32x8 z8 = { 0, 0, 0, 0, 0, 0, 0, 0 };
  __builtin_amdgcn_tensor_load_to_lds(g0, g1, z4, z4, z8, 0);
#else
  (void)ldsOff; (void)gptr;
#endif
}

// ---------------- prep kernels ----------------

__global__ __launch_bounds__(256) void cvt_bf16(const float* __restrict__ src,
                                                __bf16* __restrict__ dst) {
  size_t i = (size_t)blockIdx.x * 256 + threadIdx.x;
  dst[i] = f2bf(src[i]);
}

// W[k][n] (f32, row-major KxN) -> Wt[n][k] (bf16) so B-fragments are row reads
__global__ __launch_bounds__(256) void wtrans(const float* __restrict__ W,
                                              __bf16* __restrict__ Wt) {
  int idx = blockIdx.x * 256 + threadIdx.x;   // < 768*768
  int k = idx / HH, n = idx % HH;
  Wt[(size_t)n * HH + k] = f2bf(W[idx]);
}

// ---------------- 128x64-tile WMMA GEMM, async double-buffered --------------
// C[M,768] = A[M,768(k)] * Wt[768(n),768(k)]^T.
// A tile staged with global_load_async_to_lds_b128 (ASYNCcnt);
// B tile staged with TDM tensor_load_to_lds (TENSORcnt) when available.
__global__ __launch_bounds__(256) void wmma_gemm(
    const __bf16* __restrict__ A, const __bf16* __restrict__ Bt,
    const float* __restrict__ bias, const float* __restrict__ residual,
    __bf16* __restrict__ outB, float* __restrict__ outF, float scale) {
  __shared__ __align__(16) __bf16 As[2][128][32];   // 16 KB
  __shared__ __align__(16) __bf16 Bs[2][64][32];    //  8 KB

  const int tid  = threadIdx.x;
  const int lane = tid & 31, wave = tid >> 5;
  const int hi = lane >> 4, r = lane & 15;
  const int bm = blockIdx.x, bn = blockIdx.y;
  // A-tile async chunks: 512 x 16B -> 2 per thread
  const int aRow0 = tid >> 2, aCh = tid & 3;        // chunks tid, tid+256
  const int aRow1 = (tid + 256) >> 2;
  // B-tile fallback chunk: 256 x 16B -> 1 per thread
  const int bRow = tid >> 2;

  const __bf16* Ab = A  + (size_t)(bm * 128) * HH;
  const __bf16* Bb = Bt + (size_t)(bn * 64)  * HH;

  const int NT = HH / 32;   // 24 k-steps
  const int PEND = HAVE_TDM ? 2 : 3;

  // prologue: tile 0 -> buffer 0
  async_ld16((uint32_t)(uintptr_t)&As[0][aRow0][aCh * 8],
             Ab + (size_t)aRow0 * HH + aCh * 8);
  async_ld16((uint32_t)(uintptr_t)&As[0][aRow1][aCh * 8],
             Ab + (size_t)aRow1 * HH + aCh * 8);
#if HAVE_TDM
  if (wave == 0) tdm_load_tile((uint32_t)(uintptr_t)&Bs[0][0][0], Bb);
#else
  async_ld16((uint32_t)(uintptr_t)&Bs[0][bRow][aCh * 8],
             Bb + (size_t)bRow * HH + aCh * 8);
#endif

  v8f acc0 = {}, acc1 = {}, acc2 = {}, acc3 = {};

  for (int i = 0; i < NT; i++) {
    const int cur = i & 1, nxt = cur ^ 1;
    const int k1 = (i + 1) * 32;
    const bool more = (i + 1) < NT;
    if (more) {
      async_ld16((uint32_t)(uintptr_t)&As[nxt][aRow0][aCh * 8],
                 Ab + (size_t)aRow0 * HH + k1 + aCh * 8);
      async_ld16((uint32_t)(uintptr_t)&As[nxt][aRow1][aCh * 8],
                 Ab + (size_t)aRow1 * HH + k1 + aCh * 8);
#if HAVE_TDM
      if (wave == 0)
        tdm_load_tile((uint32_t)(uintptr_t)&Bs[nxt][0][0], Bb + k1);
#else
      async_ld16((uint32_t)(uintptr_t)&Bs[nxt][bRow][aCh * 8],
                 Bb + (size_t)bRow * HH + k1 + aCh * 8);
#endif
    }
    wait_async(more ? PEND : 0);
#if HAVE_TDM
    if (wave == 0) {
      if (more) __builtin_amdgcn_s_wait_tensorcnt(1);
      else      __builtin_amdgcn_s_wait_tensorcnt(0);
    }
#endif
    __syncthreads();   // tile i fully in LDS for all waves

    v16bf a  = ldfrag(&As[cur][0][0], 32, wave * 16 + r, 0, hi);
    v16bf b0 = ldfrag(&Bs[cur][0][0], 32,  0 + r, 0, hi);
    v16bf b1 = ldfrag(&Bs[cur][0][0], 32, 16 + r, 0, hi);
    v16bf b2 = ldfrag(&Bs[cur][0][0], 32, 32 + r, 0, hi);
    v16bf b3 = ldfrag(&Bs[cur][0][0], 32, 48 + r, 0, hi);
    acc0 = __builtin_amdgcn_wmma_f32_16x16x32_bf16(false, a, false, b0,
                                                   (short)0, acc0, false, false);
    acc1 = __builtin_amdgcn_wmma_f32_16x16x32_bf16(false, a, false, b1,
                                                   (short)0, acc1, false, false);
    acc2 = __builtin_amdgcn_wmma_f32_16x16x32_bf16(false, a, false, b2,
                                                   (short)0, acc2, false, false);
    acc3 = __builtin_amdgcn_wmma_f32_16x16x32_bf16(false, a, false, b3,
                                                   (short)0, acc3, false, false);

    __syncthreads();   // all reads done before buffer 'cur' is overwritten
  }

  // epilogue: wave owns rows [wave*16, wave*16+16), cols bn*64 + {0..63}
  v8f* accs[4] = { &acc0, &acc1, &acc2, &acc3 };
#pragma unroll
  for (int nt = 0; nt < 4; nt++) {
    v8f acc = *accs[nt];
#pragma unroll
    for (int k = 0; k < 8; k++) {
      int row = bm * 128 + wave * 16 + hi * 8 + k;
      int col = bn * 64 + nt * 16 + r;
      size_t o = (size_t)row * HH + col;
      float v = acc[k] + bias[col];
      if (outB) outB[o] = f2bf(v * scale);
      else      outF[o] = v + residual[o];
    }
  }
}

// ---------------- attention: one block per (b, h, 32-row q-tile) ------------
__global__ __launch_bounds__(256) void attn_kernel(
    const __bf16* __restrict__ Qg, const __bf16* __restrict__ Kg,
    const __bf16* __restrict__ Vg, const float* __restrict__ am,
    __bf16* __restrict__ ctx) {
  __shared__ __align__(16) __bf16 Qs[32][64];        //  4 KB (pre-scaled Q)
  __shared__ __align__(16) __bf16 kv[64 * 224];      // 28 KB: K[208][64] then Vt[64][224]
  __shared__ __align__(16) float  Sc[32][224];       // 28 KB scores/probs
  __shared__ float Mk[224];

  const int tid  = threadIdx.x;
  const int lane = tid & 31, wave = tid >> 5;
  const int hi = lane >> 4, r = lane & 15;
  const int b = blockIdx.z, h = blockIdx.y, q0 = blockIdx.x * 32;
  const size_t headOff = (size_t)b * SS * HH + (size_t)h * HDD;

  for (int j = tid; j < 224; j += 256) Mk[j] = (j < SS) ? am[b * SS + j] : 0.f;

  // Q tile (zero-padded past S)
  {
    int row = tid >> 3, ch = tid & 7;
    int s = q0 + row;
    uint4 z = {};
    uint4 val = (s < SS)
        ? *(const uint4*)(Qg + headOff + (size_t)s * HH + ch * 8) : z;
    *(uint4*)&Qs[row][ch * 8] = val;
  }
  // K rows 0..207 (rows >= 200 zeroed), row-major stride 64
  for (int idx = tid; idx < 208 * 8; idx += 256) {
    int row = idx >> 3, ch = idx & 7;
    uint4 z = {};
    uint4 val = (row < SS)
        ? *(const uint4*)(Kg + headOff + (size_t)row * HH + ch * 8) : z;
    *(uint4*)&kv[row * 64 + ch * 8] = val;
  }
  __syncthreads();

  // scores: 2 (mt) x 13 (nt) C tiles over 8 waves; K-dim = 64 (2 wmma steps)
  for (int t = wave; t < 26; t += 8) {
    int mt = t & 1, nt = t >> 1;
    v8f acc = {};
#pragma unroll
    for (int kb = 0; kb < 64; kb += 32) {
      v16bf a  = ldfrag(&Qs[0][0], 64, mt * 16 + r, kb, hi);
      v16bf bf = ldfrag(kv,        64, nt * 16 + r, kb, hi);
      acc = __builtin_amdgcn_wmma_f32_16x16x32_bf16(false, a, false, bf,
                                                    (short)0, acc, false, false);
    }
#pragma unroll
    for (int i = 0; i < 8; i++)
      Sc[mt * 16 + hi * 8 + i][nt * 16 + r] = acc[i];
  }
  __syncthreads();

  // masked softmax: 8 threads per row, width-8 shuffles (wave32)
  {
    int row = tid >> 3, j0 = tid & 7;
    float mq = (q0 + row < SS) ? am[b * SS + q0 + row] : 0.f;
    float mx = -3.4e38f;
    for (int j = j0; j < SS; j += 8) {
      float v = (mq * Mk[j] != 0.f) ? Sc[row][j] : -1e9f;
      mx = fmaxf(mx, v);
    }
    for (int o = 4; o > 0; o >>= 1) mx = fmaxf(mx, __shfl_xor(mx, o, 8));
    float sum = 0.f;
    for (int j = j0; j < SS; j += 8) {
      float v = (mq * Mk[j] != 0.f) ? Sc[row][j] : -1e9f;
      sum += __expf(v - mx);
    }
    for (int o = 4; o > 0; o >>= 1) sum += __shfl_xor(sum, o, 8);
    float inv = 1.f / sum;
    for (int j = j0; j < 224; j += 8) {
      float p = 0.f;
      if (j < SS) {
        float v = (mq * Mk[j] != 0.f) ? Sc[row][j] : -1e9f;
        p = __expf(v - mx) * inv;
      }
      Sc[row][j] = p;
    }
  }
  __syncthreads();

  // V loaded transposed: kv[d*224 + j] = V[j][d]  (j >= 200 zeroed)
  {
    int d = tid & 63, jb = tid >> 6;   // 4 j-blocks of 56
    for (int j = jb * 56; j < jb * 56 + 56; j++)
      kv[d * 224 + j] =
          (j < SS) ? Vg[headOff + (size_t)j * HH + d] : f2bf(0.f);
  }
  __syncthreads();

  // O = P @ V : 2 (mt) x 4 (nt) tiles over 8 waves; K-dim = 224 (7 steps)
  {
    int mt = wave & 1, nt = wave >> 1;
    v8f acc = {};
    for (int kb = 0; kb < 224; kb += 32) {
      v16bf a  = ldfrag_f32(&Sc[0][0], 224, mt * 16 + r, kb, hi);
      v16bf bf = ldfrag(kv, 224, nt * 16 + r, kb, hi);
      acc = __builtin_amdgcn_wmma_f32_16x16x32_bf16(false, a, false, bf,
                                                    (short)0, acc, false, false);
    }
#pragma unroll
    for (int i = 0; i < 8; i++) {
      int s = q0 + mt * 16 + hi * 8 + i;
      if (s < SS)
        ctx[headOff + (size_t)s * HH + nt * 16 + r] = f2bf(acc[i]);
    }
  }
}

// ---------------- row layernorm ----------------
__global__ __launch_bounds__(256) void ln_kernel(
    const float* __restrict__ X, const float* __restrict__ gamma,
    const float* __restrict__ beta, float* __restrict__ out) {
  __shared__ float rs[8], rss[8];
  int row = blockIdx.x, t = threadIdx.x;
  const float* x = X + (size_t)row * HH;
  float v0 = x[t], v1 = x[t + 256], v2 = x[t + 512];
  float s = v0 + v1 + v2, ss = v0 * v0 + v1 * v1 + v2 * v2;
  for (int off = 16; off > 0; off >>= 1) {
    s  += __shfl_down(s,  off, 32);
    ss += __shfl_down(ss, off, 32);
  }
  if ((t & 31) == 0) { rs[t >> 5] = s; rss[t >> 5] = ss; }
  __syncthreads();
  s = 0.f; ss = 0.f;
#pragma unroll
  for (int i = 0; i < 8; i++) { s += rs[i]; ss += rss[i]; }
  float mu  = s * (1.f / HH);
  float var = ss * (1.f / HH) - mu * mu;
  float inv = rsqrtf(var + 1e-12f);
  float* o = out + (size_t)row * HH;
  o[t]       = (v0 - mu) * inv * gamma[t]       + beta[t];
  o[t + 256] = (v1 - mu) * inv * gamma[t + 256] + beta[t + 256];
  o[t + 512] = (v2 - mu) * inv * gamma[t + 512] + beta[t + 512];
}

// ---------------- launcher ----------------
extern "C" void kernel_launch(void* const* d_in, const int* in_sizes, int n_in,
                              void* d_out, int out_size, void* d_ws,
                              size_t ws_size, hipStream_t stream) {
  (void)in_sizes; (void)n_in; (void)out_size; (void)ws_size;
  const float* hidden = (const float*)d_in[0];
  const float* am     = (const float*)d_in[1];
  const float* Wq = (const float*)d_in[2];
  const float* bq = (const float*)d_in[3];
  const float* Wk = (const float*)d_in[4];
  const float* bk = (const float*)d_in[5];
  const float* Wv = (const float*)d_in[6];
  const float* bv = (const float*)d_in[7];
  const float* Wd = (const float*)d_in[8];
  const float* bd = (const float*)d_in[9];
  const float* gamma = (const float*)d_in[10];
  const float* beta  = (const float*)d_in[11];
  float* out = (float*)d_out;

  const size_t MH = (size_t)MM * HH;
  char* p = (char*)d_ws;
  __bf16* hb = (__bf16*)p; p += MH * 2;
  __bf16* qb = (__bf16*)p; p += MH * 2;
  __bf16* kb = (__bf16*)p; p += MH * 2;
  __bf16* vb = (__bf16*)p; p += MH * 2;
  __bf16* cb = (__bf16*)p; p += MH * 2;
  float*  X  = (float*)p;  p += MH * 4;
  __bf16* Wtq = (__bf16*)p; p += (size_t)HH * HH * 2;
  __bf16* Wtk = (__bf16*)p; p += (size_t)HH * HH * 2;
  __bf16* Wtv = (__bf16*)p; p += (size_t)HH * HH * 2;
  __bf16* Wtd = (__bf16*)p; p += (size_t)HH * HH * 2;

  cvt_bf16<<<(unsigned)(MH / 256), 256, 0, stream>>>(hidden, hb);
  wtrans<<<(HH * HH) / 256, 256, 0, stream>>>(Wq, Wtq);
  wtrans<<<(HH * HH) / 256, 256, 0, stream>>>(Wk, Wtk);
  wtrans<<<(HH * HH) / 256, 256, 0, stream>>>(Wv, Wtv);
  wtrans<<<(HH * HH) / 256, 256, 0, stream>>>(Wd, Wtd);

  dim3 g(MM / 128, HH / 64);
  // Q gets the 1/sqrt(HD) fold
  wmma_gemm<<<g, 256, 0, stream>>>(hb, Wtq, bq, nullptr, qb, nullptr, 0.125f);
  wmma_gemm<<<g, 256, 0, stream>>>(hb, Wtk, bk, nullptr, kb, nullptr, 1.0f);
  wmma_gemm<<<g, 256, 0, stream>>>(hb, Wtv, bv, nullptr, vb, nullptr, 1.0f);

  attn_kernel<<<dim3(7, NHH, BB), 256, 0, stream>>>(qb, kb, vb, am, cb);

  wmma_gemm<<<g, 256, 0, stream>>>(cb, Wtd, bd, hidden, nullptr, X, 1.0f);
  ln_kernel<<<MM, 256, 0, stream>>>(X, gamma, beta, out);
}
